// MultiHeadAttentionBlock_52312701666147
// MI455X (gfx1250) — matrix operand order repaired and checked by reading, not compiled, not run on hardware
//
#include <hip/hip_runtime.h>

#define S_LEN 4096
#define NB 2
#define DMODEL 512
#define NH 8
#define DK 64
#define MTOT (NB * S_LEN) // 8192

typedef __attribute__((ext_vector_type(16))) _Float16 v16h;
typedef __attribute__((ext_vector_type(8)))  _Float16 v8h;
typedef __attribute__((ext_vector_type(8)))  float    v8f;

union V16 { v16h v; v8h h[2]; };

__device__ __forceinline__ v16h ld16(const _Float16* p) {
  V16 u;
  u.h[0] = *(const v8h*)p;
  u.h[1] = *(const v8h*)(p + 8);
  return u.v;
}
__device__ __forceinline__ v16h ld8x2(const _Float16* p0, const _Float16* p1) {
  V16 u;
  u.h[0] = *(const v8h*)p0;
  u.h[1] = *(const v8h*)p1;
  return u.v;
}
__device__ __forceinline__ v8f wmma16(v16h a, v16h b, v8f c) {
  // D = A(16x32 f16) * B(32x16 f16) + C(16x16 f32)
  return __builtin_amdgcn_wmma_f32_16x16x32_f16(false, a, false, b, (short)0, c,
                                                false, false);
}

// xor-butterfly across the 16-lane rows via v_permlane16_b32 (VALU, not LDS).
// Select nibbles: source lane for lane i (within row of 16) = i ^ k.
__device__ __forceinline__ float perm16f(float x, unsigned lo, unsigned hi) {
  int i = __float_as_int(x);
  int r = __builtin_amdgcn_permlane16(i, i, lo, hi, false, false);
  return __int_as_float(r);
}
__device__ __forceinline__ float row_max16(float x) {
  x = fmaxf(x, perm16f(x, 0x67452301u, 0xEFCDAB89u)); // xor 1
  x = fmaxf(x, perm16f(x, 0x45670123u, 0xCDEF89ABu)); // xor 3
  x = fmaxf(x, perm16f(x, 0x01234567u, 0x89ABCDEFu)); // xor 7
  x = fmaxf(x, perm16f(x, 0xFEDCBA98u, 0x76543210u)); // xor 8
  return x;
}
__device__ __forceinline__ float row_sum16(float x) {
  x += perm16f(x, 0x67452301u, 0xEFCDAB89u);
  x += perm16f(x, 0x45670123u, 0xCDEF89ABu);
  x += perm16f(x, 0x01234567u, 0x89ABCDEFu);
  x += perm16f(x, 0xFEDCBA98u, 0x76543210u);
  return x;
}

// Async 16-byte global -> LDS copy, tracked by ASYNCcnt.
__device__ __forceinline__ void async_cp16(unsigned lds_off, const _Float16* g) {
  asm volatile("global_load_async_to_lds_b128 %0, %1, off"
               :: "v"(lds_off), "v"((unsigned long long)(uintptr_t)g)
               : "memory");
}
__device__ __forceinline__ unsigned lds_off(const void* p) {
  return (unsigned)(uintptr_t)p; // generic LDS-aperture addr[31:0] = LDS offset
}

// ---------------------------------------------------------------------------
// f32 -> f16 conversion (weights)
// ---------------------------------------------------------------------------
__global__ void cvt_f16_kernel(const float* __restrict__ src,
                               _Float16* __restrict__ dst, int n) {
  int i = blockIdx.x * blockDim.x + threadIdx.x;
  if (i < n) dst[i] = (_Float16)src[i];
}

// ---------------------------------------------------------------------------
// Projection GEMM: dst = scale * (X @ W^T + b), X f32 [MTOT, 512],
// W f16 [512,512] (row n = W[n][*], contiguous in k).
// One wave -> 16 rows x 64 cols. Block = 4 waves = 16 x 256 cols.
// VT=false: write f16 [B,H,S,DK]; VT=true: write f16 [B,H,DK,S] (for V).
// ---------------------------------------------------------------------------
template <bool VT>
__global__ __launch_bounds__(128) void proj_kernel(
    const float* __restrict__ X, const _Float16* __restrict__ W,
    const float* __restrict__ bias, _Float16* __restrict__ dst, float scale) {
  const int lane = threadIdx.x & 31;
  const int wave = threadIdx.x >> 5;
  const int nl = lane & 15;
  const int hi = lane >> 4;
  const int m0 = blockIdx.y << 4;
  const int nbase = (blockIdx.x << 8) + (wave << 6);

  const float* arow = X + (size_t)(m0 + nl) * DMODEL;
  v8f acc[4] = {};

  for (int k = 0; k < DMODEL; k += 32) {
    v16h a;
    const int kk = k + hi * 8;
#pragma unroll
    for (int e = 0; e < 8; ++e) {
      a[e]     = (_Float16)arow[kk + e];
      a[e + 8] = (_Float16)arow[kk + 16 + e];
    }
    const int kb = k + hi * 8;
#pragma unroll
    for (int nt = 0; nt < 4; ++nt) {
      v16h bf = ld8x2(W + (size_t)(nbase + nt * 16 + nl) * DMODEL + kb, W + (size_t)(nbase + nt * 16 + nl) * DMODEL + kb + 16);
      acc[nt] = wmma16(a, bf, acc[nt]);
    }
  }

#pragma unroll
  for (int nt = 0; nt < 4; ++nt) {
    const int col = nbase + nt * 16 + nl;
    const float bb = bias[col];
    const int h = col >> 6, d = col & 63;
#pragma unroll
    for (int v = 0; v < 8; ++v) {
      const int m = m0 + v + 8 * hi;
      const float val = (acc[nt][v] + bb) * scale;
      const int b = m >> 12, s = m & (S_LEN - 1);
      if (!VT)
        dst[((size_t)(b * NH + h) * S_LEN + s) * DK + d] = (_Float16)val;
      else
        dst[((size_t)(b * NH + h) * DK + d) * S_LEN + s] = (_Float16)val;
    }
  }
}

// ---------------------------------------------------------------------------
// Flash attention. One wave per 16-row Q tile of one (b,h); 4 waves/block
// share async-staged K/V chunks of 64 key columns in LDS, triple-buffered so
// a single __syncthreads per iteration suffices. Q pre-scaled by 1/sqrt(DK).
// K in [B,H,S,DK]; V transposed [B,H,DK,S]. P is per-wave in LDS (in-wave
// DS ordering makes the store -> transposed-reload safe without a barrier).
// ---------------------------------------------------------------------------
#define TCHUNK 64
#define NCHUNK (S_LEN / TCHUNK) // 64

__global__ __launch_bounds__(128) void attn_kernel(
    const _Float16* __restrict__ Qh, const _Float16* __restrict__ Kh,
    const _Float16* __restrict__ Vt, _Float16* __restrict__ Ctx) {
  // KV[buf][0] = K tile [t=64][d=64]; KV[buf][1] = V tile [d=64][t=64]
  __shared__ __align__(64) _Float16 KV[3][2][TCHUNK * DK];
  __shared__ __align__(64) _Float16 Pbuf[4][16 * TCHUNK];

  const int tid = threadIdx.x;
  const int lane = tid & 31;
  const int wave = tid >> 5;
  const int nl = lane & 15;
  const int hi = lane >> 4;
  const int bh = blockIdx.y;
  const int s0 = (blockIdx.x * 4 + wave) << 4;
  const int b = bh >> 3, h = bh & 7;

  const _Float16* Qb = Qh + (size_t)bh * S_LEN * DK;
  const _Float16* Kb = Kh + (size_t)bh * S_LEN * DK;
  const _Float16* Vb = Vt + (size_t)bh * DK * S_LEN;

  // Q A-fragments for d-chunks [0,32) and [32,64), kept resident.
  const _Float16* qrow = Qb + (size_t)(s0 + nl) * DK;
  v16h qf[2];
  qf[0] = ld8x2(qrow + hi * 8,      qrow + 16 + hi * 8);
  qf[1] = ld8x2(qrow + 32 + hi * 8, qrow + 48 + hi * 8);

  v8f acc[4] = {};
  float mrow[8], lrow[8];
#pragma unroll
  for (int v = 0; v < 8; ++v) { mrow[v] = -1e30f; lrow[v] = 0.f; }

  // Issue the async copies for chunk `ci` into buffer `buf`.
  // 128 lanes x 4 x 16B = 8KB each for K and V (8 async ops per lane).
  auto issue = [&](int ci, int buf) {
    const int t0 = ci * TCHUNK;
#pragma unroll
    for (int r = 0; r < 4; ++r) {
      const int c = tid + r * 128; // 16B chunk index, 0..511
      // K tile: linear copy of rows t0..t0+63 (64 halves each)
      async_cp16(lds_off(&KV[buf][0][c * 8]), Kb + (size_t)t0 * DK + c * 8);
      // V tile [d][t]: d = c>>3, t offset = (c&7)*8
      async_cp16(lds_off(&KV[buf][1][c * 8]),
                 Vb + (size_t)(c >> 3) * S_LEN + t0 + (c & 7) * 8);
    }
  };

  issue(0, 0);

  for (int ci = 0; ci < NCHUNK; ++ci) {
    const int cur = ci % 3;
    if (ci + 1 < NCHUNK) {
      issue(ci + 1, (ci + 1) % 3);
      asm volatile("s_wait_asynccnt 8" ::: "memory"); // prev chunk landed
    } else {
      asm volatile("s_wait_asynccnt 0" ::: "memory");
    }
    __syncthreads(); // all waves: `cur` buffer complete; protects buf reuse

    const _Float16* ldsK = &KV[cur][0][0]; // [t][d]
    const _Float16* ldsV = &KV[cur][1][0]; // [d][t]

    // scores for key tiles jt: cols [t0+16*jt, t0+16*jt+16)
    v8f sc[4] = {};
#pragma unroll
    for (int jd = 0; jd < 2; ++jd) {
      const v16h qa = qf[jd];
#pragma unroll
      for (int jt = 0; jt < 4; ++jt) {
        const v16h kf = ld8x2(ldsK + (jt * 16 + nl) * DK + jd * 32 + hi * 8, ldsK + (jt * 16 + nl) * DK + jd * 32 + 16 + hi * 8);
        sc[jt] = wmma16(qa, kf, sc[jt]); // same A for 4 consecutive WMMAs
      }
    }

    _Float16* Pw = &Pbuf[wave][0]; // 16x64 f16 tile, private to this wave
    float alpha[8];
#pragma unroll
    for (int v = 0; v < 8; ++v) {
      // row r = v + 8*hi lives across the 16 lanes of this half-group
      float mx = fmaxf(fmaxf(sc[0][v], sc[1][v]), fmaxf(sc[2][v], sc[3][v]));
      mx = row_max16(mx);
      const float mnew = fmaxf(mrow[v], mx);
      float p[4];
      float rs = 0.f;
#pragma unroll
      for (int jt = 0; jt < 4; ++jt) {
        p[jt] = __expf(sc[jt][v] - mnew);
        rs += p[jt];
      }
      rs = row_sum16(rs);
      const float al = __expf(mrow[v] - mnew);
      lrow[v] = lrow[v] * al + rs;
      mrow[v] = mnew;
      alpha[v] = al;
      const int r = v + 8 * hi;
#pragma unroll
      for (int jt = 0; jt < 4; ++jt)
        Pw[r * TCHUNK + jt * 16 + nl] = (_Float16)p[jt];
    }
#pragma unroll
    for (int nt = 0; nt < 4; ++nt)
#pragma unroll
      for (int v = 0; v < 8; ++v) acc[nt][v] *= alpha[v];

    // reload P in A-fragment layout (same wave -> DS in-order, no barrier)
#pragma unroll
    for (int tc = 0; tc < 2; ++tc) {
      const v16h pf = ld8x2(Pw + nl * TCHUNK + tc * 32 + hi * 8,
                            Pw + nl * TCHUNK + tc * 32 + 16 + hi * 8);
#pragma unroll
      for (int nt = 0; nt < 4; ++nt) {
        const v16h vf =
            ld8x2(ldsV + (nt * 16 + nl) * TCHUNK + tc * 32 + hi * 8, ldsV + (nt * 16 + nl) * TCHUNK + tc * 32 + 16 + hi * 8);
        acc[nt] = wmma16(pf, vf, acc[nt]);
      }
    }
  }

  float inv[8];
#pragma unroll
  for (int v = 0; v < 8; ++v) inv[v] = 1.0f / lrow[v];
#pragma unroll
  for (int nt = 0; nt < 4; ++nt) {
#pragma unroll
    for (int v = 0; v < 8; ++v) {
      const int srow = s0 + v + 8 * hi;
      const int col = h * DK + nt * 16 + nl;
      Ctx[(size_t)(b * S_LEN + srow) * DMODEL + col] =
          (_Float16)(acc[nt][v] * inv[v]);
    }
  }
}

// ---------------------------------------------------------------------------
// Output GEMM: out(f32) = Ctx(f16) @ Wo^T + b_o
// ---------------------------------------------------------------------------
__global__ __launch_bounds__(128) void oproj_kernel(
    const _Float16* __restrict__ Xh, const _Float16* __restrict__ W,
    const float* __restrict__ bias, float* __restrict__ out) {
  const int lane = threadIdx.x & 31;
  const int wave = threadIdx.x >> 5;
  const int nl = lane & 15;
  const int hi = lane >> 4;
  const int m0 = blockIdx.y << 4;
  const int nbase = (blockIdx.x << 8) + (wave << 6);

  const _Float16* arow = Xh + (size_t)(m0 + nl) * DMODEL;
  v8f acc[4] = {};
  for (int k = 0; k < DMODEL; k += 32) {
    const v16h a = ld8x2(arow + k + hi * 8, arow + k + 16 + hi * 8);
    const int kb = k + hi * 8;
#pragma unroll
    for (int nt = 0; nt < 4; ++nt) {
      const v16h bf = ld8x2(W + (size_t)(nbase + nt * 16 + nl) * DMODEL + kb, W + (size_t)(nbase + nt * 16 + nl) * DMODEL + kb + 16);
      acc[nt] = wmma16(a, bf, acc[nt]);
    }
  }
#pragma unroll
  for (int nt = 0; nt < 4; ++nt) {
    const int col = nbase + nt * 16 + nl;
    const float bb = bias[col];
#pragma unroll
    for (int v = 0; v < 8; ++v) {
      const int m = m0 + v + 8 * hi;
      out[(size_t)m * DMODEL + col] = acc[nt][v] + bb;
    }
  }
}

// ---------------------------------------------------------------------------
extern "C" void kernel_launch(void* const* d_in, const int* in_sizes, int n_in,
                              void* d_out, int out_size, void* d_ws,
                              size_t ws_size, hipStream_t stream) {
  (void)in_sizes; (void)n_in; (void)out_size; (void)ws_size;
  const float* q  = (const float*)d_in[0];
  const float* k  = (const float*)d_in[1];
  const float* v  = (const float*)d_in[2];
  const float* wq = (const float*)d_in[3];
  const float* bq = (const float*)d_in[4];
  const float* wk = (const float*)d_in[5];
  const float* bk = (const float*)d_in[6];
  const float* wv = (const float*)d_in[7];
  const float* bv = (const float*)d_in[8];
  const float* wo = (const float*)d_in[9];
  const float* bo = (const float*)d_in[10];
  float* out = (float*)d_out;

  char* ws = (char*)d_ws;
  const size_t WSZ = (size_t)DMODEL * DMODEL * sizeof(_Float16); // 512 KB
  const size_t ASZ = (size_t)MTOT * DMODEL * sizeof(_Float16);   // 8 MB
  _Float16* Wq16 = (_Float16*)(ws);
  _Float16* Wk16 = (_Float16*)(ws + WSZ);
  _Float16* Wv16 = (_Float16*)(ws + 2 * WSZ);
  _Float16* Wo16 = (_Float16*)(ws + 3 * WSZ);
  _Float16* Qh   = (_Float16*)(ws + 4 * WSZ);           // [B,H,S,DK], pre-scaled
  _Float16* Kh   = (_Float16*)(ws + 4 * WSZ + ASZ);     // [B,H,S,DK]
  _Float16* Vt   = (_Float16*)(ws + 4 * WSZ + 2 * ASZ); // [B,H,DK,S]
  _Float16* Ctx  = (_Float16*)(ws + 4 * WSZ + 3 * ASZ); // [B,S,D] merged heads

  const int wn = DMODEL * DMODEL;
  cvt_f16_kernel<<<dim3((wn + 255) / 256), 256, 0, stream>>>(wq, Wq16, wn);
  cvt_f16_kernel<<<dim3((wn + 255) / 256), 256, 0, stream>>>(wk, Wk16, wn);
  cvt_f16_kernel<<<dim3((wn + 255) / 256), 256, 0, stream>>>(wv, Wv16, wn);
  cvt_f16_kernel<<<dim3((wn + 255) / 256), 256, 0, stream>>>(wo, Wo16, wn);

  dim3 g(DMODEL / 256, MTOT / 16); // (2, 512), 128 threads = 4 waves
  proj_kernel<false><<<g, 128, 0, stream>>>(q, Wq16, bq, Qh, 0.125f);
  proj_kernel<false><<<g, 128, 0, stream>>>(k, Wk16, bk, Kh, 1.0f);
  proj_kernel<true ><<<g, 128, 0, stream>>>(v, Wv16, bv, Vt, 1.0f);

  attn_kernel<<<dim3(S_LEN / 64, NB * NH), 128, 0, stream>>>(Qh, Kh, Vt, Ctx);

  oproj_kernel<<<g, 128, 0, stream>>>(Ctx, Wo16, bo, out);
}
